// HomogeneousTiles_64029372448828
// MI455X (gfx1250) — matrix operand order repaired and checked by reading, not compiled, not run on hardware
//
#include <hip/hip_runtime.h>
#include <math.h>

// ---------------------------------------------------------------------------
// HomogeneousTiles on MI455X (gfx1250)
//
// Pipeline (all kernels on `stream`):
//  1. blocksums : 3x4096x4096 f32 -> grayscale(-0.5 shift) -> per-8x8-block
//                 sum / sum-of-squares (f32 accum, stored f16). This is the
//                 only bandwidth-heavy pass (192 MiB ~ 8.6us @ 23.3 TB/s).
//  2. rowpass   : R = B * W  (W = 512x512 banded 0/1, 8-wide box) via
//                 v_wmma_f32_16x16x32_f16, ONE wmma per 16x16 tile per matrix.
//  3. colpass   : T = W^T * R via wmma, then per-tile std + homogeneity mask.
//  4-6. ordered compaction (count / scan / compact) -> homog[] index list.
//  7. select    : JAX threefry2x32 randint(key(seed), 64, 0, count).
//  8. gather    : assemble 64 patches of 3x64x64 -> 3x512x512 output.
// ---------------------------------------------------------------------------

typedef __attribute__((ext_vector_type(16))) _Float16 v16h;
typedef __attribute__((ext_vector_type(8)))  float    v8f;
typedef _Float16 half_t;

#define IMGW   4096
#define BG     512                 // 512x512 grid of 8x8 pixel blocks
#define NT     505                 // 505x505 tile positions (stride 8, tile 64)
#define TCOUNT (NT * NT)           // 255025
#define NWG    ((TCOUNT + 255) / 256)  // 997 compaction workgroups

// ------------------------- kernel 1: block sums ----------------------------
__global__ void ht_blocksums(const float* __restrict__ img,
                             half_t* __restrict__ bs1, half_t* __restrict__ bs2) {
  int b = blockIdx.x * 256 + threadIdx.x;          // 262144 blocks
  int by = b >> 9, bx = b & 511;
  const float* Rp = img;
  const float* Gp = img + (size_t)IMGW * IMGW;
  const float* Bp = img + 2 * (size_t)IMGW * IMGW;
  float s1 = 0.f, s2 = 0.f;
  size_t base0 = (size_t)(by * 8) * IMGW + (size_t)(bx * 8);
  #pragma unroll
  for (int y = 0; y < 8; ++y) {
    size_t o = base0 + (size_t)y * IMGW;
    const float4* r4 = (const float4*)(Rp + o);
    const float4* g4 = (const float4*)(Gp + o);
    const float4* b4 = (const float4*)(Bp + o);
    #pragma unroll
    for (int q = 0; q < 2; ++q) {
      float4 rv = r4[q], gv = g4[q], bv = b4[q];
      float gr;
      gr = fmaf(0.299f, rv.x, fmaf(0.587f, gv.x, 0.114f * bv.x)) - 0.5f; s1 += gr; s2 = fmaf(gr, gr, s2);
      gr = fmaf(0.299f, rv.y, fmaf(0.587f, gv.y, 0.114f * bv.y)) - 0.5f; s1 += gr; s2 = fmaf(gr, gr, s2);
      gr = fmaf(0.299f, rv.z, fmaf(0.587f, gv.z, 0.114f * bv.z)) - 0.5f; s1 += gr; s2 = fmaf(gr, gr, s2);
      gr = fmaf(0.299f, rv.w, fmaf(0.587f, gv.w, 0.114f * bv.w)) - 0.5f; s1 += gr; s2 = fmaf(gr, gr, s2);
    }
  }
  bs1[b] = (half_t)s1;
  bs2[b] = (half_t)s2;
}

// --------------------- kernel 2: row pass (WMMA) ---------------------------
// R[i][j] = sum_{d=0..7} B[i][j+d]  computed as 16x16 tiles: D = A(16x32) x W(32x16)
__global__ void ht_rowpass(const half_t* __restrict__ bs1, const half_t* __restrict__ bs2,
                           half_t* __restrict__ rs1, half_t* __restrict__ rs2) {
  int wave = (blockIdx.x * blockDim.x + threadIdx.x) >> 5;   // 0..1023
  int lane = threadIdx.x & 31;
  int ti = wave >> 5, tj = wave & 31;
  int n  = lane & 15;        // A row (M), W column, D column
  int hi = lane >> 4;        // lane half
  int row = ti * 16 + n;
  int j0  = tj * 16;
  v16h a1, a2, bw;
  #pragma unroll
  for (int v = 0; v < 8; ++v) {
    #pragma unroll
    for (int p = 0; p < 2; ++p) {
      int idx = v * 2 + p;
      // 16-bit A-matrix 16x32 layout (ISA 7.12.2)
      int KA = ((v >> 2) << 4) + (hi << 3) + ((v & 3) << 1) + p;
      int col = j0 + KA;
      half_t h1 = bs1[row * BG + (col & (BG - 1))];
      half_t h2 = bs2[row * BG + (col & (BG - 1))];
      a1[idx] = (col < BG) ? h1 : (half_t)0.0f;
      a2[idx] = (col < BG) ? h2 : (half_t)0.0f;
      // 16-bit B-matrix 32x16: same K order as A
      int KB = ((v >> 2) << 4) + (hi << 3) + ((v & 3) << 1) + p;
      int d  = KB - n;
      bw[idx] = (d >= 0 && d <= 7) ? (half_t)1.0f : (half_t)0.0f;
    }
  }
  v8f cz = {};
  v8f d1 = __builtin_amdgcn_wmma_f32_16x16x32_f16(false, a1, false, bw, (short)0, cz, false, false);
  v8f d2 = __builtin_amdgcn_wmma_f32_16x16x32_f16(false, a2, false, bw, (short)0, cz, false, false);
  #pragma unroll
  for (int r = 0; r < 8; ++r) {
    int orow = ti * 16 + r + (hi << 3);      // D: M = r + hi*8
    int ocol = j0 + n;                        // D: N = lane&15
    rs1[orow * BG + ocol] = (half_t)d1[r];
    rs2[orow * BG + ocol] = (half_t)d2[r];
  }
}

// --------------- kernel 3: column pass (WMMA) + std + mask -----------------
// T[i][j] = sum_{d=0..7} R[i+d][j]  as D = W^T(16x32) x R(32x16); then std/mask.
__global__ void ht_colpass(const half_t* __restrict__ rs1, const half_t* __restrict__ rs2,
                           unsigned char* __restrict__ mask) {
  int wave = (blockIdx.x * blockDim.x + threadIdx.x) >> 5;
  int lane = threadIdx.x & 31;
  int ti = wave >> 5, tj = wave & 31;
  int n  = lane & 15;
  int hi = lane >> 4;
  int i0 = ti * 16, j0 = tj * 16;
  v16h aw, b1, b2;
  #pragma unroll
  for (int v = 0; v < 8; ++v) {
    #pragma unroll
    for (int p = 0; p < 2; ++p) {
      int idx = v * 2 + p;
      int KA = ((v >> 2) << 4) + (hi << 3) + ((v & 3) << 1) + p;
      int d  = KA - n;                        // A[m][K] = 1 iff K-m in [0,7]
      aw[idx] = (d >= 0 && d <= 7) ? (half_t)1.0f : (half_t)0.0f;
      int KB = ((v >> 2) << 4) + (hi << 3) + ((v & 3) << 1) + p;
      int krow = i0 + KB;
      half_t h1 = rs1[(krow & (BG - 1)) * BG + j0 + n];
      half_t h2 = rs2[(krow & (BG - 1)) * BG + j0 + n];
      b1[idx] = (krow < BG) ? h1 : (half_t)0.0f;
      b2[idx] = (krow < BG) ? h2 : (half_t)0.0f;
    }
  }
  v8f cz = {};
  v8f d1 = __builtin_amdgcn_wmma_f32_16x16x32_f16(false, aw, false, b1, (short)0, cz, false, false);
  v8f d2 = __builtin_amdgcn_wmma_f32_16x16x32_f16(false, aw, false, b2, (short)0, cz, false, false);
  #pragma unroll
  for (int r = 0; r < 8; ++r) {
    int i = i0 + r + (hi << 3);
    int j = j0 + n;
    if (i < NT && j < NT) {
      float s1 = d1[r], s2 = d2[r];
      // variance is invariant under the -0.5 grayscale shift
      float var = (s2 - s1 * s1 * (1.0f / 4096.0f)) * (1.0f / 4096.0f);
      float sd = sqrtf(fmaxf(var, 0.0f));
      mask[i * NT + j] = (sd >= 0.005f && sd <= 0.02f) ? 1 : 0;
    }
  }
}

// ------------------ kernels 4-6: ordered compaction ------------------------
__global__ void ht_wgcount(const unsigned char* __restrict__ mask, int* __restrict__ wgcnt) {
  __shared__ int c;
  if (threadIdx.x == 0) c = 0;
  __syncthreads();
  int i = blockIdx.x * 256 + threadIdx.x;
  if (i < TCOUNT && mask[i]) atomicAdd(&c, 1);
  __syncthreads();
  if (threadIdx.x == 0) wgcnt[blockIdx.x] = c;
}

__global__ void ht_scan(const int* __restrict__ wgcnt, int* __restrict__ wgoff) {
  if (blockIdx.x == 0 && threadIdx.x == 0) {
    int run = 0;
    for (int w = 0; w < NWG; ++w) { wgoff[w] = run; run += wgcnt[w]; }
    wgoff[NWG] = run;  // total count
  }
}

__global__ void ht_compact(const unsigned char* __restrict__ mask,
                           const int* __restrict__ wgoff, int* __restrict__ homog) {
  if (threadIdx.x == 0) {
    int run = wgoff[blockIdx.x];
    int base = blockIdx.x * 256;
    for (int k = 0; k < 256; ++k) {
      int i = base + k;
      if (i < TCOUNT && mask[i]) homog[run++] = i;
    }
  }
}

// ------------------ kernel 7: threefry2x32 selection -----------------------
__device__ __forceinline__ unsigned ht_rotl(unsigned v, int n) {
  return (v << n) | (v >> (32 - n));
}

__device__ void ht_threefry(unsigned x0, unsigned x1, unsigned k0, unsigned k1,
                            unsigned* y0, unsigned* y1) {
  const int RA[4] = {13, 15, 26, 6};
  const int RB[4] = {17, 29, 16, 24};
  unsigned k2 = k0 ^ k1 ^ 0x1BD11BDAu;
  x0 += k0; x1 += k1;
  #pragma unroll
  for (int i = 0; i < 4; ++i) { x0 += x1; x1 = ht_rotl(x1, RA[i]); x1 ^= x0; }
  x0 += k1; x1 += k2 + 1u;
  #pragma unroll
  for (int i = 0; i < 4; ++i) { x0 += x1; x1 = ht_rotl(x1, RB[i]); x1 ^= x0; }
  x0 += k2; x1 += k0 + 2u;
  #pragma unroll
  for (int i = 0; i < 4; ++i) { x0 += x1; x1 = ht_rotl(x1, RA[i]); x1 ^= x0; }
  x0 += k0; x1 += k1 + 3u;
  #pragma unroll
  for (int i = 0; i < 4; ++i) { x0 += x1; x1 = ht_rotl(x1, RB[i]); x1 ^= x0; }
  x0 += k1; x1 += k2 + 4u;
  #pragma unroll
  for (int i = 0; i < 4; ++i) { x0 += x1; x1 = ht_rotl(x1, RA[i]); x1 ^= x0; }
  x0 += k2; x1 += k0 + 5u;
  *y0 = x0; *y1 = x1;
}

__global__ void ht_select(const int* __restrict__ seedp, const int* __restrict__ wgoff,
                          const int* __restrict__ homog, int* __restrict__ coords) {
  if (blockIdx.x != 0 || threadIdx.x != 0) return;
  int total = wgoff[NWG];
  unsigned span = (unsigned)(total > 0 ? total : 1);     // count = max(sum(mask),1)
  unsigned key0 = 0u, key1 = (unsigned)seedp[0];         // jax.random.key(seed)
  // split(key): counts = iota(4); blocks (0,2) and (1,3); keys = reshape(out,(2,2))
  unsigned b0a, b0b, b1a, b1b;
  ht_threefry(0u, 2u, key0, key1, &b0a, &b0b);
  ht_threefry(1u, 3u, key0, key1, &b1a, &b1b);
  unsigned k1a = b0a, k1b = b1a;   // higher-bits key
  unsigned k2a = b0b, k2b = b1b;   // lower-bits key
  unsigned mult = 65536u % span;
  mult = (unsigned)(((unsigned long long)mult * mult) % span);
  for (int p = 0; p < 64; ++p) {
    // random_bits(key,32,(64,)): blocks (i, i+32), out[i]=y0 (i<32), out[32+i]=y1
    unsigned c0 = (unsigned)(p & 31), c1 = c0 + 32u;
    unsigned y0, y1;
    ht_threefry(c0, c1, k1a, k1b, &y0, &y1);
    unsigned hb = (p < 32) ? y0 : y1;
    ht_threefry(c0, c1, k2a, k2b, &y0, &y1);
    unsigned lb = (p < 32) ? y0 : y1;
    unsigned off = (unsigned)((((unsigned long long)(hb % span)) * mult + (unsigned long long)(lb % span)) % span);
    int sel = (total > 0) ? homog[off] : 0;
    coords[p]      = (sel / NT) * 8;   // ph
    coords[64 + p] = (sel % NT) * 8;   // pw
  }
}

// ----------------------- kernel 8: gather patches --------------------------
__global__ void ht_gather(const float* __restrict__ img, const int* __restrict__ coords,
                          float* __restrict__ out) {
  int idx = blockIdx.x * 256 + threadIdx.x;     // 786432 outputs
  int c   = idx >> 18;                          // / (512*512)
  int rem = idx & 262143;
  int I = rem >> 9, J = rem & 511;
  int gi = I >> 6, y = I & 63, gj = J >> 6, x = J & 63;
  int p = gi * 8 + gj;
  int ph = coords[p], pw = coords[64 + p];
  out[idx] = img[(size_t)c * IMGW * IMGW + (size_t)(ph + y) * IMGW + (size_t)(pw + x)];
}

// ---------------------------------------------------------------------------
extern "C" void kernel_launch(void* const* d_in, const int* in_sizes, int n_in,
                              void* d_out, int out_size, void* d_ws, size_t ws_size,
                              hipStream_t stream) {
  (void)in_sizes; (void)n_in; (void)out_size; (void)ws_size;
  const float* img  = (const float*)d_in[0];
  const int*   seed = (const int*)d_in[1];
  float*       out  = (float*)d_out;
  char*        ws   = (char*)d_ws;

  half_t*        bs1    = (half_t*)(ws + 0);          // 512 KiB
  half_t*        bs2    = (half_t*)(ws + 524288);     // 512 KiB
  half_t*        rs1    = (half_t*)(ws + 1048576);    // 512 KiB
  half_t*        rs2    = (half_t*)(ws + 1572864);    // 512 KiB
  unsigned char* mask   = (unsigned char*)(ws + 2097152); // 256 KiB
  int*           wgcnt  = (int*)(ws + 2359296);       // 4 KiB
  int*           wgoff  = (int*)(ws + 2363392);       // ~4 KiB
  int*           homog  = (int*)(ws + 2368000);       // ~1 MiB
  int*           coords = (int*)(ws + 3388416);       // 512 B

  ht_blocksums<<<1024, 256, 0, stream>>>(img, bs1, bs2);
  ht_rowpass  <<<128,  256, 0, stream>>>(bs1, bs2, rs1, rs2);
  ht_colpass  <<<128,  256, 0, stream>>>(rs1, rs2, mask);
  ht_wgcount  <<<NWG,  256, 0, stream>>>(mask, wgcnt);
  ht_scan     <<<1,    32,  0, stream>>>(wgcnt, wgoff);
  ht_compact  <<<NWG,  32,  0, stream>>>(mask, wgoff, homog);
  ht_select   <<<1,    32,  0, stream>>>(seed, wgoff, homog, coords);
  ht_gather   <<<3072, 256, 0, stream>>>(img, coords, out);
}
